// CurricularNCE_loss_47416438948199
// MI455X (gfx1250) — matrix operand order repaired and checked by reading, not compiled, not run on hardware
//
#include <hip/hip_runtime.h>

typedef __attribute__((ext_vector_type(16))) _Float16 v16h;
typedef __attribute__((ext_vector_type(8)))  float    v8f;
typedef __attribute__((ext_vector_type(4)))  unsigned u32x4;
typedef __attribute__((ext_vector_type(8)))  int      i32x8;
typedef __attribute__((ext_vector_type(4)))  int      i32x4;

namespace {
constexpr int   kN    = 8192;
constexpr int   kC    = 128;
constexpr int   kNLab = 1024;
constexpr int   kMaxC = 64;

constexpr float kKeep   = 1.0f;
constexpr float kInvT   = 1.0f / 0.07f;
constexpr float kCosM   = 0.87758256189037276f;   // cos(0.5)
constexpr float kSinM   = 0.47942553860420301f;   // sin(0.5)
constexpr float kThresh = -0.87758256189037276f;  // cos(pi-0.5)
constexpr float kMM     = 0.23971276930210151f;   // sin(pi-0.5)*0.5
constexpr float kAlpha  = 0.1f;
constexpr float kValid  = 0.8f;
constexpr float kEps    = 1e-12f;

// LDS row stride in f16 elements: 128 data + 8 pad (TDM pads 4 dwords / 64 dwords)
constexpr int kLdsStride = 136;

// workspace byte offsets
constexpr size_t OFF_F1H = 0;                          // N*C f16
constexpr size_t OFF_F2H = OFF_F1H + (size_t)kN * kC * 2;
constexpr size_t OFF_CNT = OFF_F2H + (size_t)kN * kC * 2;  // kNLab i32
constexpr size_t OFF_IDX = OFF_CNT + (size_t)kNLab * 4;    // kNLab*kMaxC i32
constexpr size_t OFF_RS2 = OFF_IDX + (size_t)kNLab * kMaxC * 4; // N f32
constexpr size_t OFF_LP  = OFF_RS2 + (size_t)kN * 4;
constexpr size_t OFF_LN  = OFF_LP  + (size_t)kN * 4;
constexpr size_t OFF_S1  = OFF_LN  + (size_t)kN * 4;       // f32
constexpr size_t OFF_PC  = OFF_S1 + 4;                     // i32
constexpr size_t OFF_END = OFF_PC + 4;
} // namespace

__device__ __forceinline__ float rsum16(float v) {
  v += __shfl_xor(v, 1, 32);
  v += __shfl_xor(v, 2, 32);
  v += __shfl_xor(v, 4, 32);
  v += __shfl_xor(v, 8, 32);
  return v;
}

__device__ __forceinline__ void atomAddF(float* p, float v) {
  unsafeAtomicAdd(p, v);   // native global_atomic_add_f32
}

// Issue one TDM descriptor: load a 64x128 f16 tile (row-major, row stride 128
// elements in memory) into LDS at lds_off, padding +4 dwords per 64-dword row
// so LDS rows land at 272B stride (bank-conflict-free for row-strided reads).
__device__ __forceinline__ void tdm_load_tile(unsigned lds_off,
                                              unsigned long long ga) {
  u32x4 g0;
  g0[0] = 1u;                                        // count=1, user mode
  g0[1] = lds_off;                                   // lds_addr
  g0[2] = (unsigned)(ga & 0xFFFFFFFFull);            // global_addr[31:0]
  g0[3] = (unsigned)((ga >> 32) & 0x1FFFFFFull)      // global_addr[56:32]
          | 0x80000000u;                             // type=2 (image)
  i32x8 g1;
  g1[0] = (int)0x07510000u;      // data_size=2B, pad_en, interval=64dw, amt=4dw
  g1[1] = (int)(128u << 16);     // tensor_dim0 = 128
  g1[2] = (int)(8192u << 16);    // tensor_dim1[15:0] = 8192
  g1[3] = (int)(128u << 16);     // tile_dim0 = 128
  g1[4] = 64;                    // tile_dim1 = 64
  g1[5] = 128;                   // tensor_dim0_stride = 128
  g1[6] = 0;
  g1[7] = 0;
  i32x4 z4 = {0, 0, 0, 0};
  i32x8 z8 = {0, 0, 0, 0, 0, 0, 0, 0};
  __builtin_amdgcn_tensor_load_to_lds(g0, g1, z4, z4, z8, 0);
}

__global__ void cnce_zero_kernel(unsigned* p, int n) {
  int i = blockIdx.x * blockDim.x + threadIdx.x;
  if (i < n) p[i] = 0u;
}

// One block (128 threads) per row: L2-normalize f1/f2 rows -> f16 copies,
// and build per-label row index lists.
__global__ void cnce_norm_kernel(const float* __restrict__ f1,
                                 const float* __restrict__ f2,
                                 const int* __restrict__ labels,
                                 _Float16* __restrict__ f1h,
                                 _Float16* __restrict__ f2h,
                                 int* __restrict__ cnt,
                                 int* __restrict__ idx) {
  int r = blockIdx.x, t = threadIdx.x;
  __shared__ float sm[128];
  float x1 = f1[(size_t)r * kC + t];
  sm[t] = x1 * x1;
  __syncthreads();
  for (int s = 64; s > 0; s >>= 1) { if (t < s) sm[t] += sm[t + s]; __syncthreads(); }
  float n1 = fmaxf(sqrtf(sm[0]), kEps);
  __syncthreads();
  float x2 = f2[(size_t)r * kC + t];
  sm[t] = x2 * x2;
  __syncthreads();
  for (int s = 64; s > 0; s >>= 1) { if (t < s) sm[t] += sm[t + s]; __syncthreads(); }
  float n2 = fmaxf(sqrtf(sm[0]), kEps);
  f1h[(size_t)r * kC + t] = (_Float16)(x1 / n1);
  f2h[(size_t)r * kC + t] = (_Float16)(x2 / n2);
  if (t == 0) {
    int lb = labels[r];
    int slot = atomicAdd(&cnt[lb], 1);
    if (slot < kMaxC) idx[lb * kMaxC + slot] = r;
  }
}

// One block per label: intra-label pairs only (~N^2/1024 of all pairs).
// Accumulates S1 = sum(sim*pos), Pc = pos.sum(), rs2[i] = sum_j sim2*pos.
__global__ void cnce_pos_kernel(const _Float16* __restrict__ f1h,
                                const _Float16* __restrict__ f2h,
                                const int* __restrict__ cnt,
                                const int* __restrict__ idx,
                                float* __restrict__ rs2,
                                float* __restrict__ S1,
                                int* __restrict__ Pc) {
  int l = blockIdx.x;
  int c = cnt[l];
  if (c > kMaxC) c = kMaxC;
  int np = c * c;
  float sacc = 0.f;
  for (int p = threadIdx.x; p < np; p += blockDim.x) {
    int a = p / c, b = p - a * c;
    int ra = idx[l * kMaxC + a], rb = idx[l * kMaxC + b];
    const _Float16* x = f1h + (size_t)ra * kC;
    const _Float16* y = f2h + (size_t)rb * kC;
    float d = 0.f;
#pragma unroll 8
    for (int k = 0; k < kC; ++k) d = fmaf((float)x[k], (float)y[k], d);
    float s = fminf(fmaxf(d, -1.f), 1.f);
    sacc += s;
    float st = sqrtf(fmaxf(1.f - s * s, 0.f));
    float fp = (s > kThresh) ? (s * kCosM - st * kSinM) : (s - kMM);
    atomAddF(&rs2[ra], fp);
  }
  __shared__ float sm[128];
  sm[threadIdx.x] = sacc;
  __syncthreads();
  for (int s = 64; s > 0; s >>= 1) { if (threadIdx.x < s) sm[threadIdx.x] += sm[threadIdx.x + s]; __syncthreads(); }
  if (threadIdx.x == 0) { atomAddF(S1, sm[0]); atomicAdd(Pc, np); }
}

// Fused WMMA GEMM + curricular epilogue + per-row LP/LN reduction.
// Block = 8 waves, 64x64 output tile. TDM stages the 64-row A and B strips
// (full K=128) into padded LDS; wave w computes row-tile (w&3) x col-tiles
// (w>>2)*32 and +16 with 8 v_wmma_f32_16x16x32_f16 chained over K.
__global__ void __launch_bounds__(256)
cnce_gemm_kernel(const _Float16* __restrict__ f1h,
                 const _Float16* __restrict__ f2h,
                 const int* __restrict__ labels,
                 const int* __restrict__ cnt,
                 const float* __restrict__ rs2,
                 const float* __restrict__ S1,
                 const int* __restrict__ Pc,
                 float* __restrict__ LP,
                 float* __restrict__ LN) {
  __shared__ __align__(16) _Float16 smem[2 * 64 * kLdsStride];
  _Float16* sA = smem;
  _Float16* sB = smem + 64 * kLdsStride;

  int tid = threadIdx.x;
  int lane = tid & 31, w = tid >> 5;
  int hi = lane >> 4, l16 = lane & 15;
  int rowbase = blockIdx.y * 64 + (w & 3) * 16;
  int colbase = blockIdx.x * 64 + (w >> 2) * 32;

  if (w == 0) {
    unsigned lbase = (unsigned)(size_t)&smem[0];
    tdm_load_tile(lbase,
                  (unsigned long long)(size_t)f1h +
                      (unsigned long long)blockIdx.y * (64ull * kC * 2));
    tdm_load_tile(lbase + (unsigned)(64 * kLdsStride * 2),
                  (unsigned long long)(size_t)f2h +
                      (unsigned long long)blockIdx.x * (64ull * kC * 2));
    __builtin_amdgcn_s_wait_tensorcnt(0);
  }
  __syncthreads();

  const _Float16* aP  = sA + (size_t)((w & 3) * 16 + l16) * kLdsStride;
  const _Float16* b0P = sB + (size_t)((w >> 2) * 32 + l16) * kLdsStride;
  const _Float16* b1P = b0P + (size_t)16 * kLdsStride;

  v8f acc0 = {};
  v8f acc1 = {};
#pragma unroll
  for (int ks = 0; ks < 4; ++ks) {
    int kb = ks * 32;
    union { v16h h; unsigned u[8]; } A, B0, B1;
#pragma unroll
    for (int v = 0; v < 8; ++v) {
      // A-matrix 16x32 f16 layout: lanes 0-15 K=0..7/16..23, lanes 16-31 +8
      int kA = kb + ((v < 4) ? 2 * v : 16 + 2 * (v - 4)) + 8 * hi;
      // B-matrix 32x16 layout: lanes 0-15 K=0..7/16..23, lanes 16-31 +8
      int kB = kb + ((v < 4) ? 2 * v : 16 + 2 * (v - 4)) + 8 * hi;
      A.u[v]  = *reinterpret_cast<const unsigned*>(aP  + kA);
      B0.u[v] = *reinterpret_cast<const unsigned*>(b0P + kB);
      B1.u[v] = *reinterpret_cast<const unsigned*>(b1P + kB);
    }
    acc0 = __builtin_amdgcn_wmma_f32_16x16x32_f16(false, A.h, false, B0.h,
                                                  (short)0, acc0, false, false);
    acc1 = __builtin_amdgcn_wmma_f32_16x16x32_f16(false, A.h, false, B1.h,
                                                  (short)0, acc1, false, false);
  }

  // uniform scalars: t = alpha * mean_ts (buffer t0 = 0)
  float s1 = *S1;
  float pc = (float)(*Pc);
  float tval = kAlpha * (s1 / pc);
  bool wv = tval > kValid;

  int lc0 = labels[colbase + l16];
  int lc1 = labels[colbase + 16 + l16];

#pragma unroll
  for (int j = 0; j < 8; ++j) {
    int row = rowbase + j + 8 * hi;   // C/D layout: vgpr j -> M = j (+8 hi half)
    int lr = labels[row];
    float tgt = rs2[row] / (float)cnt[lr];
    float lp = 0.f, ln = 0.f;
#pragma unroll
    for (int tI = 0; tI < 2; ++tI) {
      float a = (tI == 0) ? acc0[j] : acc1[j];
      int lc = (tI == 0) ? lc0 : lc1;
      float s = fminf(fmaxf(a, -1.f), 1.f);
      bool pos = (lr == lc);
      float st = sqrtf(fmaxf(1.f - s * s, 0.f));
      float fp = (s > kThresh) ? (s * kCosM - st * kSinM) : (s - kMM);
      // pos entries: sim4 = final_pos; neg entries: hard-mask curricular scale
      float s4 = pos ? fp : ((s > tgt) ? s * (tval + s) : s);
      float e = __expf(s4 * kInvT);
      if (pos) {
        float d = 1.f - s;
        lp = fmaf(e, wv ? (d * d + kKeep) : 1.f, lp);
      } else {
        float sc = fminf(fmaxf(s, 0.f), 1.f);
        ln = fmaf(e, wv ? (sc * sc + kKeep) : 1.f, ln);
      }
    }
    lp = rsum16(lp);
    ln = rsum16(ln);
    if (l16 == j) {
      atomAddF(&LP[row], lp);
      atomAddF(&LN[row], ln);
    }
  }
}

__global__ void cnce_loss_kernel(const float* __restrict__ LP,
                                 const float* __restrict__ LN,
                                 float* __restrict__ out) {
  __shared__ float sm[256];
  float acc = 0.f;
  for (int i = threadIdx.x; i < kN; i += 256)
    acc += logf(1.0f + LN[i] / LP[i]);   // -log(LP/(LP+LN))
  sm[threadIdx.x] = acc;
  __syncthreads();
  for (int s = 128; s > 0; s >>= 1) { if (threadIdx.x < s) sm[threadIdx.x] += sm[threadIdx.x + s]; __syncthreads(); }
  if (threadIdx.x == 0) out[0] = sm[0] / (float)kN;
}

extern "C" void kernel_launch(void* const* d_in, const int* in_sizes, int n_in,
                              void* d_out, int out_size, void* d_ws, size_t ws_size,
                              hipStream_t stream) {
  (void)in_sizes; (void)n_in; (void)out_size; (void)ws_size;
  const float* f1 = (const float*)d_in[0];
  const float* f2 = (const float*)d_in[1];
  const int* labels = (const int*)d_in[2];

  char* ws = (char*)d_ws;
  _Float16* f1h = (_Float16*)(ws + OFF_F1H);
  _Float16* f2h = (_Float16*)(ws + OFF_F2H);
  int*   cnt = (int*)(ws + OFF_CNT);
  int*   idx = (int*)(ws + OFF_IDX);
  float* rs2 = (float*)(ws + OFF_RS2);
  float* LP  = (float*)(ws + OFF_LP);
  float* LN  = (float*)(ws + OFF_LN);
  float* S1  = (float*)(ws + OFF_S1);
  int*   Pc  = (int*)(ws + OFF_PC);

  int zwords = (int)((OFF_END - OFF_CNT) / 4);
  cnce_zero_kernel<<<(zwords + 255) / 256, 256, 0, stream>>>(
      (unsigned*)(ws + OFF_CNT), zwords);

  cnce_norm_kernel<<<kN, kC, 0, stream>>>(f1, f2, labels, f1h, f2h, cnt, idx);

  cnce_pos_kernel<<<kNLab, 128, 0, stream>>>(f1h, f2h, cnt, idx, rs2, S1, Pc);

  dim3 grid(kN / 64, kN / 64);
  cnce_gemm_kernel<<<grid, 256, 0, stream>>>(f1h, f2h, labels, cnt, rs2, S1, Pc,
                                             LP, LN);

  cnce_loss_kernel<<<1, 256, 0, stream>>>(LP, LN, (float*)d_out);
}
